// AudioCompressedLayer_40681930228298
// MI455X (gfx1250) — matrix operand order repaired and checked by reading, not compiled, not run on hardware
//
#include <hip/hip_runtime.h>
#include <hip/hip_bf16.h>

// ---------------------------------------------------------------------------
// Types for CDNA5 WMMA
// ---------------------------------------------------------------------------
typedef __attribute__((ext_vector_type(16))) __bf16 v16bf;
typedef __attribute__((ext_vector_type(8)))  float  v8f;

#define FFT_N    2048
#define FFT_LOG  11
#define N_WIN    8192            // 4096*4096 / 2048 windows
#define IN_F     4096
#define OUT_F    4096
#define BATCH    2048
#define ENERGY_THRESH 0.01f
#define KEEP_FRAC     0.7f       // 1 - COMPRESSION_RATIO

// ---------------------------------------------------------------------------
// init: zero stats buffers + norm
// ---------------------------------------------------------------------------
__global__ void k_init(float* freq_e, float* time_e, unsigned* normbits) {
    int i = blockIdx.x * blockDim.x + threadIdx.x;
    if (i < FFT_N) freq_e[i] = 0.0f;
    if (i < N_WIN) time_e[i] = 0.0f;
    if (i == 0)    *normbits = 0u;
}

// ---------------------------------------------------------------------------
// max |w| reduction (positive-float bit-pattern atomicMax)
// ---------------------------------------------------------------------------
__global__ __launch_bounds__(256) void k_absmax(const float* __restrict__ w,
                                                unsigned* normbits, int n) {
    __shared__ float red[256];
    float m = 0.0f;
    for (int i = blockIdx.x * blockDim.x + threadIdx.x; i < n;
         i += gridDim.x * blockDim.x)
        m = fmaxf(m, fabsf(w[i]));
    red[threadIdx.x] = m;
    __syncthreads();
    for (int off = 128; off > 0; off >>= 1) {
        if (threadIdx.x < off)
            red[threadIdx.x] = fmaxf(red[threadIdx.x], red[threadIdx.x + off]);
        __syncthreads();
    }
    if (threadIdx.x == 0) atomicMax(normbits, __float_as_uint(red[0]));
}

// ---------------------------------------------------------------------------
// shared-memory radix-2 FFT stages (input must be bit-reversed); sgn=-1 fwd
// ---------------------------------------------------------------------------
__device__ __forceinline__ void fft_stages(float* re, float* im, float sgn) {
    for (int s = 1; s <= FFT_LOG; ++s) {
        const int half = 1 << (s - 1);
        const int m    = 1 << s;
        for (int b = threadIdx.x; b < FFT_N / 2; b += blockDim.x) {
            int grp = b >> (s - 1);
            int pos = b & (half - 1);
            int i0  = grp * m + pos;
            int i1  = i0 + half;
            float ang = sgn * 6.2831853071795864769f * (float)pos / (float)m;
            float c = __cosf(ang), sn = __sinf(ang);
            float xr = re[i1], xi = im[i1];
            float tr = c * xr - sn * xi;
            float ti = c * xi + sn * xr;
            re[i1] = re[i0] - tr;  im[i1] = im[i0] - ti;
            re[i0] += tr;          im[i0] += ti;
        }
        __syncthreads();
    }
}

// ---------------------------------------------------------------------------
// per-window forward FFT + energy accumulation
// ---------------------------------------------------------------------------
__global__ __launch_bounds__(512) void k_fft_energy(const float* __restrict__ w,
                                                    const unsigned* normbits,
                                                    float* freq_e, float* time_e) {
    __shared__ float re[FFT_N], im[FFT_N];
    __shared__ float red[512];
    const int t = blockIdx.x;
    float norm = __uint_as_float(*normbits);
    float inv  = (norm > 0.0f) ? 1.0f / norm : 1.0f;

    for (int i = threadIdx.x; i < FFT_N; i += blockDim.x) {
        int j = __brev((unsigned)i) >> (32 - FFT_LOG);
        re[j] = w[(size_t)t * FFT_N + i] * inv;
        im[j] = 0.0f;
    }
    __syncthreads();
    fft_stages(re, im, -1.0f);

    float local = 0.0f;
    for (int i = threadIdx.x; i < FFT_N; i += blockDim.x) {
        float e = re[i] * re[i] + im[i] * im[i];
        atomicAdd(&freq_e[i], e * (1.0f / (float)N_WIN));
        local += e;
    }
    red[threadIdx.x] = local;
    __syncthreads();
    for (int off = 256; off > 0; off >>= 1) {
        if (threadIdx.x < off) red[threadIdx.x] += red[threadIdx.x + off];
        __syncthreads();
    }
    if (threadIdx.x == 0) time_e[t] = red[0] * (1.0f / (float)FFT_N);
}

// ---------------------------------------------------------------------------
// top-k mask via single-workgroup bitonic sort (n = 2048 or 8192, pow2)
// ---------------------------------------------------------------------------
__global__ __launch_bounds__(1024) void k_mask(const float* __restrict__ e,
                                               float* __restrict__ mask, int n) {
    __shared__ float s[N_WIN];       // up to 8192 floats = 32 KB
    __shared__ float red[1024];
    __shared__ float sh_thresh, sh_kth;
    __shared__ int   sh_count, sh_target;
    const int tid = threadIdx.x, bd = blockDim.x;

    for (int i = tid; i < n; i += bd) s[i] = e[i];
    __syncthreads();

    // max
    float m = 0.0f;
    for (int i = tid; i < n; i += bd) m = fmaxf(m, s[i]);
    red[tid] = m; __syncthreads();
    for (int off = 512; off > 0; off >>= 1) {
        if (tid < off) red[tid] = fmaxf(red[tid], red[tid + off]);
        __syncthreads();
    }
    if (tid == 0) sh_thresh = ENERGY_THRESH * red[0];
    __syncthreads();
    const float thresh = sh_thresh;

    // count above threshold
    float c = 0.0f;
    for (int i = tid; i < n; i += bd) c += (s[i] > thresh) ? 1.0f : 0.0f;
    red[tid] = c; __syncthreads();
    for (int off = 512; off > 0; off >>= 1) {
        if (tid < off) red[tid] += red[tid + off];
        __syncthreads();
    }
    if (tid == 0) {
        int cnt = (int)(red[0] + 0.5f);
        int tgt = (int)floorf((float)cnt * KEEP_FRAC);
        sh_count  = cnt;
        sh_target = (tgt < 1) ? 1 : tgt;
    }
    __syncthreads();
    const int count = sh_count, target = sh_target;

    // bitonic sort ascending
    for (int k2 = 2; k2 <= n; k2 <<= 1) {
        for (int j = k2 >> 1; j > 0; j >>= 1) {
            for (int i = tid; i < n; i += bd) {
                int l = i ^ j;
                if (l > i) {
                    bool up = ((i & k2) == 0);
                    float a = s[i], b = s[l];
                    if ((a > b) == up) { s[i] = b; s[l] = a; }
                }
            }
            __syncthreads();
        }
    }
    if (tid == 0) sh_kth = s[n - target];
    __syncthreads();
    const float kth = sh_kth;

    for (int i = tid; i < n; i += bd) {
        float v  = e[i];
        float mk = (target < count) ? (v >= kth ? 1.0f : 0.0f)
                                    : (v > thresh ? 1.0f : 0.0f);
        mask[i] = mk;
    }
}

// ---------------------------------------------------------------------------
// recompute FFT, apply outer-product mask, inverse FFT, write bf16 weight
// ---------------------------------------------------------------------------
__global__ __launch_bounds__(512) void k_reconstruct(const float* __restrict__ w,
                                                     const unsigned* normbits,
                                                     const float* __restrict__ fmask,
                                                     const float* __restrict__ tmask,
                                                     __hip_bfloat16* __restrict__ wbf) {
    __shared__ float re[FFT_N], im[FFT_N];
    const int t = blockIdx.x;
    float norm = __uint_as_float(*normbits);
    if (!(norm > 0.0f)) norm = 1.0f;
    const float inv = 1.0f / norm;
    const float tm  = tmask[t];

    for (int i = threadIdx.x; i < FFT_N; i += blockDim.x) {
        int j = __brev((unsigned)i) >> (32 - FFT_LOG);
        re[j] = w[(size_t)t * FFT_N + i] * inv;
        im[j] = 0.0f;
    }
    __syncthreads();
    fft_stages(re, im, -1.0f);           // forward FFT, natural order out

    for (int i = threadIdx.x; i < FFT_N; i += blockDim.x) {
        float f = tm * fmask[i];
        re[i] *= f;  im[i] *= f;
    }
    __syncthreads();

    // in-place bit reversal before inverse stages
    for (int i = threadIdx.x; i < FFT_N; i += blockDim.x) {
        int j = __brev((unsigned)i) >> (32 - FFT_LOG);
        if (i < j) {
            float tr = re[i]; re[i] = re[j]; re[j] = tr;
            float ti = im[i]; im[i] = im[j]; im[j] = ti;
        }
    }
    __syncthreads();
    fft_stages(re, im, +1.0f);           // inverse stages

    const float scale = norm * (1.0f / (float)FFT_N);
    for (int i = threadIdx.x; i < FFT_N; i += blockDim.x)
        wbf[(size_t)t * FFT_N + i] = __float2bfloat16(re[i] * scale);
}

// ---------------------------------------------------------------------------
// x (f32) -> bf16
// ---------------------------------------------------------------------------
__global__ __launch_bounds__(256) void k_xcvt(const float* __restrict__ x,
                                              __hip_bfloat16* __restrict__ xbf, int n) {
    int i = (blockIdx.x * blockDim.x + threadIdx.x) * 4;
    if (i + 3 < n) {
        float4 v = *(const float4*)(x + i);
        xbf[i + 0] = __float2bfloat16(v.x);
        xbf[i + 1] = __float2bfloat16(v.y);
        xbf[i + 2] = __float2bfloat16(v.z);
        xbf[i + 3] = __float2bfloat16(v.w);
    }
}

// ---------------------------------------------------------------------------
// WMMA bf16 GEMM: out[m,n] = sum_k X[m,k]*W[n,k] + bias[n]
// block = 256 threads = 8 waves (2x4); block tile 128(M) x 256(N);
// wave tile 64 x 64 = 4x4 WMMA accumulators. The 16 WMMAs are issued in a
// rotated (mt, (mt+j)&3) order so ALL 8 fragments stay live across the whole
// WMMA block -> compiler hoists all 16 b128 loads and uses staggered partial
// s_wait_loadcnt instead of full drains between 4-WMMA groups.
// ---------------------------------------------------------------------------
__device__ __forceinline__ v16bf load_fragA(const __hip_bfloat16* __restrict__ base,
                                            int row, int kc, int lane) {
    // ISA 16-bit A 16x32 layout: lanes 0-15 hold K = [0..7]+[16..23],
    // lanes 16-31 hold K = [8..15]+[24..31]
    const int r  = row + (lane & 15);
    const int kh = (lane >> 4) * 8;
    const __hip_bfloat16* p = base + (size_t)r * IN_F + kc + kh;
    union { uint4 u[2]; v16bf v; } u;
    u.u[0] = *(const uint4*)(p);
    u.u[1] = *(const uint4*)(p + 16);
    return u.v;
}

__device__ __forceinline__ v16bf load_fragB(const __hip_bfloat16* __restrict__ base,
                                            int row, int kc, int lane) {
    // B 32x16 layout (matches A): lanes 0-15 hold K = [0..7]+[16..23], lanes 16-31 K = [8..15]+[24..31]
    const int r  = row + (lane & 15);
    const int kh = (lane >> 4) * 8;
    const __hip_bfloat16* p = base + (size_t)r * IN_F + kc + kh;
    union { uint4 u[2]; v16bf v; } u;
    u.u[0] = *(const uint4*)(p);
    u.u[1] = *(const uint4*)(p + 16);
    return u.v;
}

__global__ __launch_bounds__(256) void k_gemm(const __hip_bfloat16* __restrict__ X,
                                              const __hip_bfloat16* __restrict__ W,
                                              const float* __restrict__ bias,
                                              float* __restrict__ out) {
    const int lane = threadIdx.x & 31;
    const int wave = threadIdx.x >> 5;          // 0..7
    const int wm   = wave >> 2;                 // 0..1
    const int wn   = wave & 3;                  // 0..3
    const int mBase = blockIdx.y * 128 + wm * 64;
    const int nBase = blockIdx.x * 256 + wn * 64;

    v8f acc[4][4];
    const v8f vzero = {0.f, 0.f, 0.f, 0.f, 0.f, 0.f, 0.f, 0.f};
#pragma unroll
    for (int mt = 0; mt < 4; ++mt)
#pragma unroll
        for (int nt = 0; nt < 4; ++nt)
            acc[mt][nt] = vzero;

#pragma unroll 1
    for (int kc = 0; kc < IN_F; kc += 32) {
        v16bf a[4], b[4];
#pragma unroll
        for (int mt = 0; mt < 4; ++mt)
            a[mt] = load_fragA(X, mBase + mt * 16, kc, lane);
#pragma unroll
        for (int nt = 0; nt < 4; ++nt)
            b[nt] = load_fragB(W, nBase + nt * 16, kc, lane);

        // rotated order: every a[] and b[] is used in every round of 4,
        // keeping all fragments live so loads can be hoisted & waits staggered
#pragma unroll
        for (int j = 0; j < 4; ++j)
#pragma unroll
            for (int mt = 0; mt < 4; ++mt) {
                const int nt = (mt + j) & 3;
                acc[mt][nt] = __builtin_amdgcn_wmma_f32_16x16x32_bf16(
                    false, a[mt], false, b[nt], (short)0, acc[mt][nt],
                    false, false);
            }
    }

    // C layout: VGPR i -> M = i + 8*(lane>>4), N = lane&15
    const int ncol = lane & 15;
    const int mhi  = (lane >> 4) * 8;
#pragma unroll
    for (int mt = 0; mt < 4; ++mt)
#pragma unroll
        for (int nt = 0; nt < 4; ++nt) {
            const int n = nBase + nt * 16 + ncol;
            const float bv = bias[n];
#pragma unroll
            for (int i = 0; i < 8; ++i) {
                const int m = mBase + mt * 16 + mhi + i;
                out[(size_t)m * OUT_F + n] = acc[mt][nt][i] + bv;
            }
        }
}

// ---------------------------------------------------------------------------
// launch
// ---------------------------------------------------------------------------
extern "C" void kernel_launch(void* const* d_in, const int* in_sizes, int n_in,
                              void* d_out, int out_size, void* d_ws, size_t ws_size,
                              hipStream_t stream) {
    const float* x    = (const float*)d_in[0];
    const float* w    = (const float*)d_in[1];
    const float* bias = (const float*)d_in[2];
    float*       out  = (float*)d_out;

    char* ws = (char*)d_ws;
    __hip_bfloat16* xbf  = (__hip_bfloat16*)(ws + 0);                 // 16 MB
    __hip_bfloat16* wbf  = (__hip_bfloat16*)(ws + 16777216);          // 32 MB
    float*   freq_e      = (float*)(ws + 50331648);                   // 8 KB
    float*   time_e      = (float*)(ws + 50339840);                   // 32 KB
    float*   fmask       = (float*)(ws + 50372608);                   // 8 KB
    float*   tmask       = (float*)(ws + 50380800);                   // 32 KB
    unsigned* normbits   = (unsigned*)(ws + 50413568);                // 4 B

    const int wN = in_sizes[1];   // 4096*4096
    const int xN = in_sizes[0];   // 2048*4096

    k_init<<<32, 256, 0, stream>>>(freq_e, time_e, normbits);
    k_absmax<<<1024, 256, 0, stream>>>(w, normbits, wN);
    k_fft_energy<<<N_WIN, 512, 0, stream>>>(w, normbits, freq_e, time_e);
    k_mask<<<1, 1024, 0, stream>>>(freq_e, fmask, FFT_N);
    k_mask<<<1, 1024, 0, stream>>>(time_e, tmask, N_WIN);
    k_reconstruct<<<N_WIN, 512, 0, stream>>>(w, normbits, fmask, tmask, wbf);
    k_xcvt<<<(xN / 4 + 255) / 256, 256, 0, stream>>>(x, xbf, xN);

    dim3 grid(OUT_F / 256, BATCH / 128);   // (16, 16)
    k_gemm<<<grid, 256, 0, stream>>>(xbf, wbf, bias, out);
}
